// RNA_ClassQuery_Model_45887430590918
// MI455X (gfx1250) — matrix operand order repaired and checked by reading, not compiled, not run on hardware
//
#include <hip/hip_runtime.h>
#include <hip/hip_bf16.h>
#include <math.h>

// ---------------------------------------------------------------------------
// Problem constants (from reference)
// ---------------------------------------------------------------------------
static constexpr int B_    = 128;
static constexpr int CIN   = 16;
static constexpr int D_    = 128;
static constexpr int C_    = 12;      // class queries
static constexpr int H_    = 4;       // heads
static constexpr int DH_   = 32;
static constexpr int EDGES = 640000;
static constexpr int N_    = 128000;  // B*NPG nodes (multiple of 128)
static constexpr int NL_   = 3;
static constexpr int FF_   = 512;     // 4*D

typedef _Float16 v4h  __attribute__((ext_vector_type(4)));
typedef _Float16 v8h  __attribute__((ext_vector_type(8)));
typedef _Float16 v16h __attribute__((ext_vector_type(16)));
typedef float    v8f  __attribute__((ext_vector_type(8)));

__device__ __forceinline__ float wave_sum32(float v) {
#pragma unroll
  for (int o = 16; o > 0; o >>= 1) v += __shfl_xor(v, o, 32);
  return v;
}

// ---------------------------------------------------------------------------
// Generic fills
// ---------------------------------------------------------------------------
__global__ void fill_f32(float* p, float v, int n) {
  int i = blockIdx.x * blockDim.x + threadIdx.x;
  if (i < n) p[i] = v;
}
__global__ void fill_i32(int* p, int v, int n) {
  int i = blockIdx.x * blockDim.x + threadIdx.x;
  if (i < n) p[i] = v;
}

// ---------------------------------------------------------------------------
// Degree / dinv / counts / starts
// ---------------------------------------------------------------------------
__global__ void deg_accum(const int* __restrict__ col, float* __restrict__ deg) {
  int e = blockIdx.x * blockDim.x + threadIdx.x;
  if (e < EDGES) atomicAdd(&deg[col[e]], 1.0f);
}
__global__ void rsqrt_inplace(float* p, int n) {
  int i = blockIdx.x * blockDim.x + threadIdx.x;
  if (i < n) p[i] = rsqrtf(p[i]);
}
__global__ void count_nodes(const int* __restrict__ batch, int* __restrict__ cnt) {
  int i = blockIdx.x * blockDim.x + threadIdx.x;
  if (i < N_) atomicAdd(&cnt[batch[i]], 1);
}
__global__ void scan_starts(const int* __restrict__ cnt, int* __restrict__ starts) {
  if (blockIdx.x == 0 && threadIdx.x == 0) {
    int s = 0;
    for (int b = 0; b < B_; ++b) { starts[b] = s; s += cnt[b]; }
  }
}

// ---------------------------------------------------------------------------
// Input projection h = x @ in_w^T + in_b   ([N,16] -> [N,128])
// ---------------------------------------------------------------------------
__global__ void input_proj(const float* __restrict__ x, const float* __restrict__ w,
                           const float* __restrict__ b, float* __restrict__ h) {
  int idx = blockIdx.x * blockDim.x + threadIdx.x;
  if (idx >= N_ * D_) return;
  int i = idx >> 7, d = idx & 127;
  const float* xr = x + i * CIN;
  const float* wr = w + d * CIN;
  float s = b[d];
#pragma unroll
  for (int c = 0; c < CIN; ++c) s = fmaf(xr[c], wr[c], s);
  h[idx] = s;
}

// ---------------------------------------------------------------------------
// WMMA GEMM:  O[M,128] = A[M,128] @ W[128,128]^T (+bias), M % 128 == 0.
// block = 256 threads (8 wave32), computes a 128x128 output patch.
// A tile + W staged in dynamic LDS as f16 (row stride 136 halves = 272B,
// 16B-aligned, bank-spread). Per wave: B fragments (4 K-steps) held in
// registers; per row-tile: 8 grouped ds_load_b128 -> one wait -> 4 WMMAs.
// ---------------------------------------------------------------------------
__global__ __launch_bounds__(256) void gemm128_wmma(
    const float* __restrict__ A, const float* __restrict__ W,
    const float* __restrict__ bias, float* __restrict__ O) {
  extern __shared__ _Float16 smem[];            // 16B aligned
  _Float16* Wl = smem;                          // [128][136]
  _Float16* Al = smem + 128 * 136;              // [128][136]

  const int tid  = threadIdx.x;
  const int base = blockIdx.x * 128;

  // stage W and the 128-row A tile as f16 (b128 global loads, b64 LDS stores)
  for (int idx = tid; idx < 128 * 32; idx += 256) {
    int n = idx >> 5, k4 = (idx & 31) * 4;
    float4 w4 = *(const float4*)(W + n * 128 + k4);
    v4h wh = { (_Float16)w4.x, (_Float16)w4.y, (_Float16)w4.z, (_Float16)w4.w };
    *(v4h*)(Wl + n * 136 + k4) = wh;
    float4 a4 = *(const float4*)(A + (base + n) * 128 + k4);
    v4h ah = { (_Float16)a4.x, (_Float16)a4.y, (_Float16)a4.z, (_Float16)a4.w };
    *(v4h*)(Al + n * 136 + k4) = ah;
  }
  __syncthreads();

  const int lane = tid & 31;
  const int ln16 = lane & 15;
  const int half = lane >> 4;
  const int n0   = (tid >> 5) * 16;

  // B fragments: column n holds K halves of W row n in the same interleaved K layout as A
  const _Float16* Wrow = Wl + (n0 + ln16) * 136;
  v16h bf[4];
#pragma unroll
  for (int ks = 0; ks < 4; ++ks) {
    v8h blo = *(const v8h*)(Wrow + ks * 32 + half * 8);
    v8h bhi = *(const v8h*)(Wrow + ks * 32 + 16 + half * 8);
    bf[ks] = __builtin_shufflevector(blo, bhi, 0,1,2,3,4,5,6,7,8,9,10,11,12,13,14,15);
  }
  const float bn = bias ? bias[n0 + ln16] : 0.0f;

#pragma unroll 1
  for (int rt = 0; rt < 8; ++rt) {
    // A fragments: interleaved K-pair layout, all 4 K-steps loaded up front
    const _Float16* Arow = Al + (rt * 16 + ln16) * 136;
    v16h af[4];
#pragma unroll
    for (int ks = 0; ks < 4; ++ks) {
      v8h alo = *(const v8h*)(Arow + ks * 32 + half * 8);
      v8h ahi = *(const v8h*)(Arow + ks * 32 + 16 + half * 8);
      af[ks] = __builtin_shufflevector(alo, ahi, 0,1,2,3,4,5,6,7,8,9,10,11,12,13,14,15);
    }
    v8f acc;
#pragma unroll
    for (int r = 0; r < 8; ++r) acc[r] = bn;
#pragma unroll
    for (int ks = 0; ks < 4; ++ks)
      acc = __builtin_amdgcn_wmma_f32_16x16x32_f16(false, af[ks], false, bf[ks],
                                                   (short)0, acc, false, false);
    float* Orow = O + (base + rt * 16) * 128 + n0 + ln16;
#pragma unroll
    for (int r = 0; r < 8; ++r)
      Orow[(r + 8 * half) * 128] = acc[r];
  }
}

// ---------------------------------------------------------------------------
// GCN edge scatter: agg[col] += dinv[row]*dinv[col] * hw[row]   (wave per edge)
// ---------------------------------------------------------------------------
__global__ void edge_scatter(const int* __restrict__ row, const int* __restrict__ col,
                             const float* __restrict__ dinv, const float* __restrict__ hw,
                             float* __restrict__ agg) {
  int t = blockIdx.x * blockDim.x + threadIdx.x;
  int e = t >> 5;
  if (e >= EDGES) return;
  int lane = t & 31;
  int r = row[e], c = col[e];
  float w = dinv[r] * dinv[c];
#pragma unroll
  for (int j = 0; j < 4; ++j) {
    int d = lane + 32 * j;
    atomicAdd(&agg[c * 128 + d], w * hw[r * 128 + d]);
  }
}

// ---------------------------------------------------------------------------
// GCN finalize (fused): v = agg + dinv^2*hw + b; LN; [relu + residual]; h = v
// one wave per node row
// ---------------------------------------------------------------------------
__global__ void gcn_finalize(float* __restrict__ h, const float* __restrict__ agg,
                             const float* __restrict__ hw, const float* __restrict__ dinv,
                             const float* __restrict__ gb, const float* __restrict__ lw,
                             const float* __restrict__ lb, int relu_residual) {
  int gw = (blockIdx.x * blockDim.x + threadIdx.x) >> 5;
  int lane = threadIdx.x & 31;
  if (gw >= N_) return;
  float sw = dinv[gw] * dinv[gw];
  float v[4], s = 0.f;
#pragma unroll
  for (int j = 0; j < 4; ++j) {
    int d = lane + 32 * j;
    v[j] = agg[gw * 128 + d] + sw * hw[gw * 128 + d] + gb[d];
    s += v[j];
  }
  float mean = wave_sum32(s) * (1.0f / 128.0f);
  float var = 0.f;
#pragma unroll
  for (int j = 0; j < 4; ++j) { float t = v[j] - mean; var += t * t; }
  var = wave_sum32(var) * (1.0f / 128.0f);
  float inv = rsqrtf(var + 1e-5f);
#pragma unroll
  for (int j = 0; j < 4; ++j) {
    int d = lane + 32 * j;
    float y = (v[j] - mean) * inv * lw[d] + lb[d];
    if (relu_residual) y = fmaxf(y, 0.0f) + h[gw * 128 + d];
    h[gw * 128 + d] = y;
  }
}

// ---------------------------------------------------------------------------
// t init: t[b,c,:] = cq[c,:]
// ---------------------------------------------------------------------------
__global__ void t_init(float* __restrict__ t, const float* __restrict__ cq) {
  int idx = blockIdx.x * blockDim.x + threadIdx.x;
  if (idx < B_ * C_ * D_) t[idx] = cq[idx % (C_ * D_)];
}

// ---------------------------------------------------------------------------
// LN of the C=12 query rows of one batch into LDS (helper, 8 waves)
// ---------------------------------------------------------------------------
__device__ __forceinline__ void ln_rows_to_lds(const float* __restrict__ tb,
                                               const float* __restrict__ w,
                                               const float* __restrict__ b,
                                               float (*out)[128]) {
  int wave = threadIdx.x >> 5, lane = threadIdx.x & 31;
  for (int c = wave; c < C_; c += 8) {
    float v[4], s = 0.f;
#pragma unroll
    for (int j = 0; j < 4; ++j) { v[j] = tb[c * 128 + lane + 32 * j]; s += v[j]; }
    float mean = wave_sum32(s) * (1.0f / 128.0f);
    float var = 0.f;
#pragma unroll
    for (int j = 0; j < 4; ++j) { float t = v[j] - mean; var += t * t; }
    var = wave_sum32(var) * (1.0f / 128.0f);
    float inv = rsqrtf(var + 1e-5f);
#pragma unroll
    for (int j = 0; j < 4; ++j) {
      int d = lane + 32 * j;
      out[c][d] = (v[j] - mean) * inv * w[d] + b[d];
    }
  }
}

// ---------------------------------------------------------------------------
// Self-attention block for one batch (12 queries, packed in_proj)
// ---------------------------------------------------------------------------
__global__ __launch_bounds__(256) void self_attn(
    float* __restrict__ t, const float* __restrict__ qkv_w, const float* __restrict__ qkv_b,
    const float* __restrict__ out_w, const float* __restrict__ out_b,
    const float* __restrict__ n1w, const float* __restrict__ n1b) {
  __shared__ float tn[C_][128];
  __shared__ float qs[C_][128];
  __shared__ float ks2[C_][128];
  __shared__ float vs[C_][128];
  __shared__ float os[C_][128];
  int b = blockIdx.x, tid = threadIdx.x;
  float* tb = t + b * C_ * D_;

  ln_rows_to_lds(tb, n1w, n1b, tn);
  __syncthreads();

  for (int idx = tid; idx < 3 * C_ * 128; idx += 256) {
    int rowi = idx >> 7, d = idx & 127;
    int part = rowi / C_, c = rowi % C_;
    int wr = part * 128 + d;
    const float* wp = qkv_w + wr * 128;
    float s = qkv_b[wr];
    for (int k = 0; k < 128; ++k) s = fmaf(tn[c][k], wp[k], s);
    (part == 0 ? qs : part == 1 ? ks2 : vs)[c][d] = s;
  }
  __syncthreads();

  if (tid < H_ * C_) {
    int hh = tid / C_, c = tid % C_;
    float sc[C_], mx = -INFINITY;
    for (int j = 0; j < C_; ++j) {
      float s = 0.f;
      for (int dd = 0; dd < DH_; ++dd) s = fmaf(qs[c][hh * 32 + dd], ks2[j][hh * 32 + dd], s);
      s *= 0.17677669529663687f;  // 1/sqrt(32)
      sc[j] = s;
      mx = fmaxf(mx, s);
    }
    float l = 0.f;
    for (int j = 0; j < C_; ++j) { sc[j] = __expf(sc[j] - mx); l += sc[j]; }
    float li = 1.0f / l;
    for (int dd = 0; dd < DH_; ++dd) {
      float a = 0.f;
      for (int j = 0; j < C_; ++j) a = fmaf(sc[j], vs[j][hh * 32 + dd], a);
      os[c][hh * 32 + dd] = a * li;
    }
  }
  __syncthreads();

  for (int idx = tid; idx < C_ * 128; idx += 256) {
    int c = idx >> 7, d = idx & 127;
    const float* wp = out_w + d * 128;
    float s = out_b[d];
    for (int k = 0; k < 128; ++k) s = fmaf(os[c][k], wp[k], s);
    tb[c * 128 + d] += s;
  }
}

// ---------------------------------------------------------------------------
// Cross-attention: query projection q2 = LN(t,n2) @ Wq^T + bq  (per batch)
// ---------------------------------------------------------------------------
__global__ __launch_bounds__(256) void cross_q(
    const float* __restrict__ t, const float* __restrict__ qkv_w,
    const float* __restrict__ qkv_b, const float* __restrict__ n2w,
    const float* __restrict__ n2b, float* __restrict__ q2) {
  __shared__ float tn[C_][128];
  int b = blockIdx.x, tid = threadIdx.x;
  ln_rows_to_lds(t + b * C_ * D_, n2w, n2b, tn);
  __syncthreads();
  for (int idx = tid; idx < C_ * 128; idx += 256) {
    int c = idx >> 7, d = idx & 127;
    const float* wp = qkv_w + d * 128;   // rows [0:D] = Q
    float s = qkv_b[d];
    for (int k = 0; k < 128; ++k) s = fmaf(tn[c][k], wp[k], s);
    q2[(b * C_ + c) * 128 + d] = s;
  }
}

// ---------------------------------------------------------------------------
// Cross-attention streaming softmax: block per (batch, head), wave per query.
// Padded keys (j >= count[b]) contribute zero weight => skipped.
// ---------------------------------------------------------------------------
__global__ __launch_bounds__(384) void cross_stream(
    const float* __restrict__ q2, const float* __restrict__ Kh,
    const float* __restrict__ Vh, const int* __restrict__ cnt,
    const int* __restrict__ starts, float* __restrict__ attno) {
  int b = blockIdx.x >> 2, hh = blockIdx.x & 3;
  int c = threadIdx.x >> 5, lane = threadIdx.x & 31;
  int n = cnt[b], st = starts[b];
  float q = q2[(b * C_ + c) * 128 + hh * 32 + lane] * 0.17677669529663687f;
  float m = -INFINITY, l = 0.f, acc = 0.f;
  for (int j = 0; j < n; ++j) {
    int roff = (st + j) * 128 + hh * 32;
    float s = wave_sum32(q * Kh[roff + lane]);
    float mn = fmaxf(m, s);
    float sc = __expf(m - mn);   // first iter: exp(-inf)=0
    float p = __expf(s - mn);
    l = l * sc + p;
    acc = acc * sc + p * Vh[roff + lane];
    m = mn;
  }
  attno[(b * C_ + c) * 128 + hh * 32 + lane] = acc / l;
}

// ---------------------------------------------------------------------------
// Cross-attention output projection + residual (per batch)
// ---------------------------------------------------------------------------
__global__ __launch_bounds__(256) void cross_out(
    float* __restrict__ t, const float* __restrict__ attno,
    const float* __restrict__ out_w, const float* __restrict__ out_b) {
  __shared__ float os[C_][128];
  int b = blockIdx.x, tid = threadIdx.x;
  for (int idx = tid; idx < C_ * 128; idx += 256)
    os[idx >> 7][idx & 127] = attno[b * C_ * 128 + idx];
  __syncthreads();
  for (int idx = tid; idx < C_ * 128; idx += 256) {
    int c = idx >> 7, d = idx & 127;
    const float* wp = out_w + d * 128;
    float s = out_b[d];
    for (int k = 0; k < 128; ++k) s = fmaf(os[c][k], wp[k], s);
    t[b * C_ * 128 + idx] += s;
  }
}

// ---------------------------------------------------------------------------
// FFN: t += relu(LN(t,n3) @ W1^T + b1) @ W2^T + b2  (per batch)
// ---------------------------------------------------------------------------
__global__ __launch_bounds__(256) void ffn(
    float* __restrict__ t, const float* __restrict__ n3w, const float* __restrict__ n3b,
    const float* __restrict__ w1, const float* __restrict__ b1,
    const float* __restrict__ w2, const float* __restrict__ b2) {
  __shared__ float tn[C_][128];
  __shared__ float hid[C_][FF_];
  int b = blockIdx.x, tid = threadIdx.x;
  ln_rows_to_lds(t + b * C_ * D_, n3w, n3b, tn);
  __syncthreads();
  for (int idx = tid; idx < C_ * FF_; idx += 256) {
    int c = idx >> 9, e = idx & 511;
    const float* wp = w1 + e * 128;
    float s = b1[e];
    for (int k = 0; k < 128; ++k) s = fmaf(tn[c][k], wp[k], s);
    hid[c][e] = fmaxf(s, 0.0f);
  }
  __syncthreads();
  for (int idx = tid; idx < C_ * 128; idx += 256) {
    int c = idx >> 7, d = idx & 127;
    const float* wp = w2 + d * FF_;
    float s = b2[d];
    for (int e = 0; e < FF_; ++e) s = fmaf(hid[c][e], wp[e], s);
    t[b * C_ * 128 + idx] += s;
  }
}

// ---------------------------------------------------------------------------
// Output: logits[b,c] = LN(t[b,c]) . op_w + op_b   (one wave per (b,c))
// ---------------------------------------------------------------------------
__global__ void logits_kernel(const float* __restrict__ t, const float* __restrict__ lnw,
                              const float* __restrict__ lnb, const float* __restrict__ opw,
                              const float* __restrict__ opb, float* __restrict__ out) {
  int gw = (blockIdx.x * blockDim.x + threadIdx.x) >> 5;
  int lane = threadIdx.x & 31;
  if (gw >= B_ * C_) return;
  const float* r = t + gw * 128;
  float v[4], s = 0.f;
#pragma unroll
  for (int j = 0; j < 4; ++j) { v[j] = r[lane + 32 * j]; s += v[j]; }
  float mean = wave_sum32(s) * (1.0f / 128.0f);
  float var = 0.f;
#pragma unroll
  for (int j = 0; j < 4; ++j) { float d = v[j] - mean; var += d * d; }
  var = wave_sum32(var) * (1.0f / 128.0f);
  float inv = rsqrtf(var + 1e-5f);
  float p = 0.f;
#pragma unroll
  for (int j = 0; j < 4; ++j) {
    int d = lane + 32 * j;
    p += ((v[j] - mean) * inv * lnw[d] + lnb[d]) * opw[d];
  }
  p = wave_sum32(p);
  if (lane == 0) out[gw] = p + opb[0];
}

// ---------------------------------------------------------------------------
// Host launcher
// ---------------------------------------------------------------------------
extern "C" void kernel_launch(void* const* d_in, const int* in_sizes, int n_in,
                              void* d_out, int out_size, void* d_ws, size_t ws_size,
                              hipStream_t stream) {
  (void)in_sizes; (void)n_in; (void)out_size; (void)ws_size;

  const float* x       = (const float*)d_in[0];
  const int*   ei      = (const int*)d_in[1];     // [2,E]
  const int*   batch   = (const int*)d_in[2];
  const float* in_w    = (const float*)d_in[3];
  const float* in_b    = (const float*)d_in[4];
  const float* gcn_w   = (const float*)d_in[5];   // [3,D,D]
  const float* gcn_b   = (const float*)d_in[6];
  const float* ln_w    = (const float*)d_in[7];
  const float* ln_b    = (const float*)d_in[8];
  const float* cq      = (const float*)d_in[9];
  const float* sa_qkvw = (const float*)d_in[10];
  const float* sa_qkvb = (const float*)d_in[11];
  const float* sa_ow   = (const float*)d_in[12];
  const float* sa_ob   = (const float*)d_in[13];
  const float* ca_qkvw = (const float*)d_in[14];
  const float* ca_qkvb = (const float*)d_in[15];
  const float* ca_ow   = (const float*)d_in[16];
  const float* ca_ob   = (const float*)d_in[17];
  const float* n1w = (const float*)d_in[18]; const float* n1b = (const float*)d_in[19];
  const float* n2w = (const float*)d_in[20]; const float* n2b = (const float*)d_in[21];
  const float* n3w = (const float*)d_in[22]; const float* n3b = (const float*)d_in[23];
  const float* ff1w = (const float*)d_in[24]; const float* ff1b = (const float*)d_in[25];
  const float* ff2w = (const float*)d_in[26]; const float* ff2b = (const float*)d_in[27];
  const float* oplnw = (const float*)d_in[28]; const float* oplnb = (const float*)d_in[29];
  const float* opw = (const float*)d_in[30]; const float* opb = (const float*)d_in[31];
  float* out = (float*)d_out;

  const int* erow = ei;
  const int* ecol = ei + EDGES;

  // ---- workspace carving (~200 MB; big buffers ping-ponged) ----
  const size_t ND = (size_t)N_ * D_;        // 16.384M floats
  const size_t BCD = (size_t)B_ * C_ * D_;  // 196,608 floats
  float* h    = (float*)d_ws;               // node features (persists)
  float* bufB = h + ND;                     // hw, then Kh
  float* bufC = bufB + ND;                  // agg, then Vh
  float* dinv = bufC + ND;                  // deg -> rsqrt in place
  float* t    = dinv + N_;                  // queries [B,C,D]
  float* q2   = t + BCD;
  float* atno = q2 + BCD;
  int* counts = (int*)(atno + BCD);
  int* starts = counts + B_;

  const int TPB = 256;
  const size_t gemm_smem = 2u * 128u * 136u * sizeof(_Float16);  // 69,632 B

  // ---- degree normalization ----
  fill_f32<<<(N_ + TPB - 1) / TPB, TPB, 0, stream>>>(dinv, 1.0f, N_);
  deg_accum<<<(EDGES + TPB - 1) / TPB, TPB, 0, stream>>>(ecol, dinv);
  rsqrt_inplace<<<(N_ + TPB - 1) / TPB, TPB, 0, stream>>>(dinv, N_);

  // ---- batch counts / starts (generic, for cross-attention) ----
  fill_i32<<<1, TPB, 0, stream>>>(counts, 0, B_);
  count_nodes<<<(N_ + TPB - 1) / TPB, TPB, 0, stream>>>(batch, counts);
  scan_starts<<<1, 32, 0, stream>>>(counts, starts);

  // ---- input projection ----
  input_proj<<<(N_ * D_ + TPB - 1) / TPB, TPB, 0, stream>>>(x, in_w, in_b, h);

  // ---- 3 GCN layers ----
  const int gemm_blocks = N_ / 128;  // 1000 (N_ % 128 == 0)
  for (int i = 0; i < NL_; ++i) {
    gemm128_wmma<<<gemm_blocks, TPB, gemm_smem, stream>>>(
        h, gcn_w + (size_t)i * D_ * D_, nullptr, bufB);
    fill_f32<<<(int)((ND + TPB - 1) / TPB), TPB, 0, stream>>>(bufC, 0.0f, (int)ND);
    edge_scatter<<<(EDGES * 32 + TPB - 1) / TPB, TPB, 0, stream>>>(erow, ecol, dinv, bufB, bufC);
    gcn_finalize<<<(N_ * 32 + TPB - 1) / TPB, TPB, 0, stream>>>(
        h, bufC, bufB, dinv, gcn_b + i * D_, ln_w + i * D_, ln_b + i * D_, i < NL_ - 1 ? 1 : 0);
  }

  // ---- cross-attention K/V projections over node features ----
  gemm128_wmma<<<gemm_blocks, TPB, gemm_smem, stream>>>(
      h, ca_qkvw + (size_t)D_ * D_, ca_qkvb + D_, bufB);         // Kh
  gemm128_wmma<<<gemm_blocks, TPB, gemm_smem, stream>>>(
      h, ca_qkvw + (size_t)2 * D_ * D_, ca_qkvb + 2 * D_, bufC); // Vh

  // ---- transformer decoder layer (norm_first) ----
  t_init<<<(int)((BCD + TPB - 1) / TPB), TPB, 0, stream>>>(t, cq);
  self_attn<<<B_, TPB, 0, stream>>>(t, sa_qkvw, sa_qkvb, sa_ow, sa_ob, n1w, n1b);
  cross_q<<<B_, TPB, 0, stream>>>(t, ca_qkvw, ca_qkvb, n2w, n2b, q2);
  cross_stream<<<B_ * H_, 384, 0, stream>>>(q2, bufB, bufC, counts, starts, atno);
  cross_out<<<B_, TPB, 0, stream>>>(t, atno, ca_ow, ca_ob);
  ffn<<<B_, TPB, 0, stream>>>(t, n3w, n3b, ff1w, ff1b, ff2w, ff2b);

  // ---- output projection ----
  logits_kernel<<<(B_ * C_ * 32 + TPB - 1) / TPB, TPB, 0, stream>>>(
      t, oplnw, oplnb, opw, opb, out);
}
